// CMAttention_22222160789862
// MI455X (gfx1250) — matrix operand order repaired and checked by reading, not compiled, not run on hardware
//
#include <hip/hip_runtime.h>

// ---------------- problem constants ----------------
#define BATCH   2
#define NTOK    2048
#define DIM     512
#define NHEAD   8
#define DHEAD   64
#define MMEM    3
#define INNER   512           // NHEAD*DHEAD
#define NPAD    2080          // 2048 + 32 (3 memory tokens + zero pad), multiple of 32
#define NVALID  2051          // NTOK + MMEM
#define SCALE   0.125f        // DHEAD^-0.5
#define SQRT_DH 8.0f
#define SQRT_M  1.7320508075688772f

typedef __attribute__((ext_vector_type(16))) __bf16 v16bf;
typedef __attribute__((ext_vector_type(8)))  __bf16 v8bf;
typedef __attribute__((ext_vector_type(8)))  float  v8f;

// ---------------- WMMA helpers ----------------
__device__ __forceinline__ v8f wmma_bf16(v16bf a, v16bf b, v8f c) {
  // 8 args: (neg_a, A, neg_b, B, c_mod, C, reuse_a, reuse_b)
  return __builtin_amdgcn_wmma_f32_16x16x32_bf16(false, a, false, b, (short)0, c,
                                                 false, false);
}

// A fragment (16x32, MxK): lane holds row (lane&15); K = base + i,
// base = (lane&16)?16:0.  Two contiguous 16B loads.
__device__ __forceinline__ v16bf load_fragA(const __bf16* base, int stride,
                                            int row, int k0, int lane) {
  const __bf16* p = base + (size_t)row * stride + k0 + ((lane & 16) ? 16 : 0);
  v8bf lo = *reinterpret_cast<const v8bf*>(p);
  v8bf hi = *reinterpret_cast<const v8bf*>(p + 8);
  return __builtin_shufflevector(lo, hi, 0, 1, 2, 3, 4, 5, 6, 7,
                                 8, 9, 10, 11, 12, 13, 14, 15);
}

// B fragment (32x16, KxN) read from B^T stored row-major (N x K):
// lane holds column (lane&15); K = ((lane&16)?16:0) + i  -> one 32B load.
__device__ __forceinline__ v16bf load_fragB(const __bf16* bt, int stride,
                                            int ncol, int k0, int lane) {
  const __bf16* p = bt + (size_t)ncol * stride + k0 + ((lane & 16) ? 16 : 0);
  return *reinterpret_cast<const v16bf*>(p);
}

// ---------------- prep kernels ----------------
__global__ void k_cvt_x(const float* __restrict__ x, __bf16* __restrict__ xb, int n) {
  int i = blockIdx.x * blockDim.x + threadIdx.x;
  if (i < n) xb[i] = (__bf16)x[i];
}

// wT[o][i] : o<512 -> wq column o ; 512<=o<1536 -> wkv column (o-512)
__global__ void k_build_w(const float* __restrict__ wq, const float* __restrict__ wkv,
                          __bf16* __restrict__ wT) {
  int i = blockIdx.x * blockDim.x + threadIdx.x;
  if (i >= (INNER * 3) * DIM) return;
  int o = i >> 9, ii = i & 511;
  float v = (o < INNER) ? wq[ii * INNER + o] : wkv[ii * (2 * INNER) + (o - INNER)];
  wT[i] = (__bf16)v;
}

__global__ void k_build_wo(const float* __restrict__ wo, __bf16* __restrict__ woT) {
  int i = blockIdx.x * blockDim.x + threadIdx.x;
  if (i >= DIM * INNER) return;
  int o = i >> 9, ii = i & 511;
  woT[i] = (__bf16)wo[ii * DIM + o];
}

// Fill K rows [2048,2080) and V^T cols [2048,2080) per (b,h):
// memory tokens use the reference's flat (M,INNER)->(H,M,DH) reshape:
// element (h, r, d) = mk_flat[h*192 + r*64 + d].
__global__ void k_fill_mem(const float* __restrict__ mk, const float* __restrict__ mv,
                           __bf16* __restrict__ kbuf, __bf16* __restrict__ vtbuf) {
  int i = blockIdx.x * blockDim.x + threadIdx.x;
  if (i >= BATCH * NHEAD * 32 * DHEAD) return;
  int d = i & 63;
  int r = (i >> 6) & 31;          // 0..31 -> pad row j = 2048+r
  int bh = i >> 11;               // 0..15
  int h = bh & 7;
  int j = NTOK + r;
  float kv = 0.f, vv = 0.f;
  if (r < MMEM) {
    int flat = h * (MMEM * DHEAD) + r * DHEAD + d;
    kv = SQRT_DH * mk[flat];
    vv = SQRT_M * mv[flat];
  }
  kbuf[((size_t)bh * NPAD + j) * DHEAD + d] = (__bf16)kv;
  vtbuf[((size_t)bh * DHEAD + d) * NPAD + j] = (__bf16)vv;
}

// ---------------- fused QKV projection GEMM ----------------
// C(4096x1536) = xb(4096x512) @ [wq|wk|wv]; 32x32 tile per wave (2x data reuse).
__global__ __launch_bounds__(256) void k_qkv_gemm(const __bf16* __restrict__ xb,
                                                  const __bf16* __restrict__ wT,
                                                  __bf16* __restrict__ qbuf,
                                                  __bf16* __restrict__ kbuf,
                                                  __bf16* __restrict__ vtbuf) {
  int lane = threadIdx.x & 31, wave = threadIdx.x >> 5;
  int t = blockIdx.x * 8 + wave;              // 0 .. 128*48-1
  int tm = t / 48, tn = t % 48;
  int mb = tm * 32, nb = tn * 32;
  int arow = lane & 15, jcol = lane & 15, rowoff = (lane >> 4) << 3;
  v8f zero = {};
  v8f acc[2][2] = {{zero, zero}, {zero, zero}};
#pragma unroll
  for (int kb = 0; kb < DIM; kb += 32) {
    v16bf a0 = load_fragA(xb, DIM, mb + arow, kb, lane);
    v16bf a1 = load_fragA(xb, DIM, mb + 16 + arow, kb, lane);
    v16bf b0 = load_fragB(wT, DIM, nb + jcol, kb, lane);
    v16bf b1 = load_fragB(wT, DIM, nb + 16 + jcol, kb, lane);
    acc[0][0] = wmma_bf16(a0, b0, acc[0][0]);
    acc[0][1] = wmma_bf16(a0, b1, acc[0][1]);
    acc[1][0] = wmma_bf16(a1, b0, acc[1][0]);
    acc[1][1] = wmma_bf16(a1, b1, acc[1][1]);
  }
#pragma unroll
  for (int mi = 0; mi < 2; ++mi) {
#pragma unroll
    for (int ni = 0; ni < 2; ++ni) {
      int col = nb + ni * 16 + jcol;
#pragma unroll
      for (int j = 0; j < 8; ++j) {
        int row = mb + mi * 16 + rowoff + j;  // global row = b*NTOK + n
        int b_ = row >> 11, n_ = row & (NTOK - 1);
        __bf16 val = (__bf16)acc[mi][ni][j];
        if (col < INNER) {
          int h = col >> 6, d = col & 63;
          qbuf[((size_t)(b_ * NHEAD + h) * NTOK + n_) * DHEAD + d] = val;
        } else if (col < 2 * INNER) {
          int o = col - INNER; int h = o >> 6, d = o & 63;
          kbuf[((size_t)(b_ * NHEAD + h) * NPAD + n_) * DHEAD + d] = val;
        } else {
          int o = col - 2 * INNER; int h = o >> 6, d = o & 63;
          vtbuf[((size_t)(b_ * NHEAD + h) * DHEAD + d) * NPAD + n_] = val;
        }
      }
    }
  }
}

// ---------------- flash attention ----------------
// One wave per 32-row q tile (two 16-row WMMA row-tiles sharing every K/V
// fragment); stream j in steps of 32 (one WMMA K-step).
__global__ __launch_bounds__(128) void k_attn(const __bf16* __restrict__ qbuf,
                                              const __bf16* __restrict__ kbuf,
                                              const __bf16* __restrict__ vtbuf,
                                              __bf16* __restrict__ aout) {
  __shared__ __bf16 smem[4][2 * 16 * 32];     // per-wave P staging (C->A relayout)
  int lane = threadIdx.x & 31, wave = threadIdx.x >> 5;
  int gw = blockIdx.x * 4 + wave;             // 0..1023
  int qt = gw & 63;                           // N/32 tiles
  int bh = gw >> 6;                           // 0..15
  const __bf16* qh = qbuf + (size_t)bh * NTOK * DHEAD;
  const __bf16* kh = kbuf + (size_t)bh * NPAD * DHEAD;
  const __bf16* vh = vtbuf + (size_t)bh * DHEAD * NPAD;
  __bf16* sp = &smem[wave][0];

  int arow = lane & 15;
  int jcol = lane & 15;
  int rowoff = (lane >> 4) << 3;

  v16bf aq[2][2];                             // [row tile][d half]
#pragma unroll
  for (int r = 0; r < 2; ++r) {
    aq[r][0] = load_fragA(qh, DHEAD, qt * 32 + r * 16 + arow, 0, lane);
    aq[r][1] = load_fragA(qh, DHEAD, qt * 32 + r * 16 + arow, 32, lane);
  }

  float mrun[2][8], lsum[2][8];
  v8f zero = {};
  v8f oacc[2][4] = {{zero, zero, zero, zero}, {zero, zero, zero, zero}};
#pragma unroll
  for (int r = 0; r < 2; ++r)
#pragma unroll
    for (int j = 0; j < 8; ++j) { mrun[r][j] = -3.0e38f; lsum[r][j] = 0.f; }

  for (int jt = 0; jt < NPAD / 32; ++jt) {
    int j0 = jt * 32;
    // S = Q K^T: 2 row tiles x 2 col subtiles, K-dim=64 -> 8 WMMAs, 4 K frags
    v8f s[2][2] = {{zero, zero}, {zero, zero}};
#pragma unroll
    for (int c = 0; c < 2; ++c) {
#pragma unroll
      for (int hh = 0; hh < 2; ++hh) {
        v16bf bk = load_fragB(kh, DHEAD, j0 + c * 16 + jcol, hh * 32, lane);
        s[0][c] = wmma_bf16(aq[0][hh], bk, s[0][c]);
        s[1][c] = wmma_bf16(aq[1][hh], bk, s[1][c]);
      }
    }
    bool v0 = (j0 + jcol) < NVALID;
    bool v1 = (j0 + 16 + jcol) < NVALID;
    // online softmax per row (rows live in 16-lane halves; reduce over lanes)
#pragma unroll
    for (int r = 0; r < 2; ++r) {
#pragma unroll
      for (int j = 0; j < 8; ++j) {
        float a0 = v0 ? s[r][0][j] * SCALE : -1.0e30f;
        float a1 = v1 ? s[r][1][j] * SCALE : -1.0e30f;
        float mx = fmaxf(a0, a1);
        mx = fmaxf(mx, __shfl_xor(mx, 1));
        mx = fmaxf(mx, __shfl_xor(mx, 2));
        mx = fmaxf(mx, __shfl_xor(mx, 4));
        mx = fmaxf(mx, __shfl_xor(mx, 8));
        float mn = fmaxf(mrun[r][j], mx);
        float p0 = __expf(a0 - mn);
        float p1 = __expf(a1 - mn);
        float corr = __expf(mrun[r][j] - mn);
        mrun[r][j] = mn;
        float rs = p0 + p1;
        rs += __shfl_xor(rs, 1);
        rs += __shfl_xor(rs, 2);
        rs += __shfl_xor(rs, 4);
        rs += __shfl_xor(rs, 8);
        lsum[r][j] = lsum[r][j] * corr + rs;
#pragma unroll
        for (int dt = 0; dt < 4; ++dt) oacc[r][dt][j] *= corr;
        sp[r * 512 + (rowoff + j) * 32 + jcol] = (__bf16)p0;
        sp[r * 512 + (rowoff + j) * 32 + 16 + jcol] = (__bf16)p1;
      }
    }
    // P (C-layout) -> A fragments via LDS; same-wave DS ops are in-order
    v16bf ap0 = load_fragA(sp, 32, arow, 0, lane);
    v16bf ap1 = load_fragA(sp + 512, 32, arow, 0, lane);
#pragma unroll
    for (int dt = 0; dt < 4; ++dt) {
      v16bf bv = load_fragB(vh, NPAD, dt * 16 + jcol, j0, lane);
      oacc[0][dt] = wmma_bf16(ap0, bv, oacc[0][dt]);
      oacc[1][dt] = wmma_bf16(ap1, bv, oacc[1][dt]);
    }
  }
  // normalize and store into (b*N+n, INNER) bf16 layout for the out-proj GEMM
  int b_ = bh >> 3, h_ = bh & 7;
#pragma unroll
  for (int r = 0; r < 2; ++r) {
#pragma unroll
    for (int j = 0; j < 8; ++j) {
      float inv = 1.0f / lsum[r][j];
      int n_ = qt * 32 + r * 16 + rowoff + j;
      size_t rowbase = ((size_t)(b_ * NTOK + n_)) * INNER + h_ * DHEAD;
#pragma unroll
      for (int dt = 0; dt < 4; ++dt)
        aout[rowbase + dt * 16 + jcol] = (__bf16)(oacc[r][dt][j] * inv);
    }
  }
}

// ---------------- output projection ----------------
__global__ __launch_bounds__(256) void k_oproj(const __bf16* __restrict__ ao,
                                               const __bf16* __restrict__ woT,
                                               const float* __restrict__ bo,
                                               float* __restrict__ out) {
  int lane = threadIdx.x & 31, wave = threadIdx.x >> 5;
  int t = blockIdx.x * 8 + wave;              // 0 .. 128*16-1
  int tm = t >> 4, tn = t & 15;
  int mb = tm * 32, nb = tn * 32;
  int arow = lane & 15, jcol = lane & 15, rowoff = (lane >> 4) << 3;
  v8f zero = {};
  v8f acc[2][2] = {{zero, zero}, {zero, zero}};
#pragma unroll
  for (int kb = 0; kb < INNER; kb += 32) {
    v16bf a0 = load_fragA(ao, INNER, mb + arow, kb, lane);
    v16bf a1 = load_fragA(ao, INNER, mb + 16 + arow, kb, lane);
    v16bf b0 = load_fragB(woT, INNER, nb + jcol, kb, lane);
    v16bf b1 = load_fragB(woT, INNER, nb + 16 + jcol, kb, lane);
    acc[0][0] = wmma_bf16(a0, b0, acc[0][0]);
    acc[0][1] = wmma_bf16(a0, b1, acc[0][1]);
    acc[1][0] = wmma_bf16(a1, b0, acc[1][0]);
    acc[1][1] = wmma_bf16(a1, b1, acc[1][1]);
  }
#pragma unroll
  for (int mi = 0; mi < 2; ++mi) {
#pragma unroll
    for (int ni = 0; ni < 2; ++ni) {
      int col = nb + ni * 16 + jcol;
      float bias = bo[col];
#pragma unroll
      for (int j = 0; j < 8; ++j) {
        int row = mb + mi * 16 + rowoff + j;
        out[(size_t)row * DIM + col] = acc[mi][ni][j] + bias;
      }
    }
  }
}

// ---------------- host launch ----------------
extern "C" void kernel_launch(void* const* d_in, const int* in_sizes, int n_in,
                              void* d_out, int out_size, void* d_ws, size_t ws_size,
                              hipStream_t stream) {
  const float* x   = (const float*)d_in[0];
  const float* wq  = (const float*)d_in[1];
  const float* wkv = (const float*)d_in[2];
  const float* wo  = (const float*)d_in[3];
  const float* bo  = (const float*)d_in[4];
  const float* m_k = (const float*)d_in[5];
  const float* m_v = (const float*)d_in[6];
  float* out = (float*)d_out;

  const size_t ROWS = (size_t)BATCH * NTOK;   // 4096
  char* ws = (char*)d_ws;
  size_t off = 0;
  auto take = [&](size_t elems) {
    char* p = ws + off;
    off = (off + elems * sizeof(__bf16) + 255) & ~(size_t)255;
    return (__bf16*)p;
  };
  __bf16* xb    = take(ROWS * DIM);                           // 4096x512
  __bf16* wT    = take((size_t)(3 * INNER) * DIM);            // 1536x512 (q|k|v cols)
  __bf16* woT   = take((size_t)DIM * INNER);                  // 512x512
  __bf16* qbuf  = take((size_t)BATCH * NHEAD * NTOK * DHEAD); // per-head Q
  __bf16* kbuf  = take((size_t)BATCH * NHEAD * NPAD * DHEAD); // per-head padded K
  __bf16* vtbuf = take((size_t)BATCH * NHEAD * DHEAD * NPAD); // per-head V^T
  __bf16* aout  = take(ROWS * INNER);                         // attention output
  (void)ws_size; (void)in_sizes; (void)n_in; (void)out_size;

  int nxe = (int)(ROWS * DIM);
  k_cvt_x<<<(nxe + 255) / 256, 256, 0, stream>>>(x, xb, nxe);
  k_build_w<<<(3 * INNER * DIM + 255) / 256, 256, 0, stream>>>(wq, wkv, wT);
  k_build_wo<<<(DIM * INNER + 255) / 256, 256, 0, stream>>>(wo, woT);
  k_fill_mem<<<(BATCH * NHEAD * 32 * DHEAD + 255) / 256, 256, 0, stream>>>(m_k, m_v, kbuf, vtbuf);

  // 4096/32 x 1536/32 = 128*48 tiles, 8 waves per 256-thread block
  k_qkv_gemm<<<(128 * 48) / 8, 256, 0, stream>>>(xb, wT, qbuf, kbuf, vtbuf);

  // B*H*(N/32) = 1024 waves, 4 waves per 128-thread block
  k_attn<<<1024 / 4, 128, 0, stream>>>(qbuf, kbuf, vtbuf, aout);

  // 4096/32 x 512/32 = 128*16 tiles
  k_oproj<<<(128 * 16) / 8, 256, 0, stream>>>(aout, woT, bo, out);
}
